// MaskedRecurrentModule_56710748176697
// MI455X (gfx1250) — matrix operand order repaired and checked by reading, not compiled, not run on hardware
//
#include <hip/hip_runtime.h>
#include <hip/hip_bf16.h>

// Problem dims (fixed by the reference)
#define T_STEPS 512
#define NENV    64
#define DIN     1024
#define HID     1024
#define G3      3072   // 3*HID

typedef __attribute__((ext_vector_type(16))) __bf16 v16bf;
typedef __attribute__((ext_vector_type(8)))  float  v8f;

struct bfrag_bits { uint4 lo, hi; };
static_assert(sizeof(bfrag_bits) == 32, "frag size");

__device__ __forceinline__ v8f wmma_bf16(v16bf a, v16bf b, v8f c) {
  // D = A(16x32 bf16) x B(32x16 bf16) + C(16x16 f32)
  return __builtin_amdgcn_wmma_f32_16x16x32_bf16(
      /*neg_a=*/false, a, /*neg_b=*/false, b,
      /*c_mod=*/(short)0, c, /*reuse_a=*/false, /*reuse_b=*/false);
}

__device__ __forceinline__ v8f zero8() {
  v8f z = {0.f, 0.f, 0.f, 0.f, 0.f, 0.f, 0.f, 0.f};
  return z;
}

// B fragment: lane holds 16 contiguous bf16 (K-major row of W), 32B at p.
__device__ __forceinline__ v16bf load_bfrag(const __bf16* p) {
  const uint4* q = reinterpret_cast<const uint4*>(p);
  bfrag_bits f; f.lo = q[0]; f.hi = q[1];
  return __builtin_bit_cast(v16bf, f);
}

// A fragment (bf16 source): lane holds K {kb..kb+15}:
// 32B contiguous at p.
__device__ __forceinline__ v16bf load_afrag(const __bf16* p) {
  const uint4* q = reinterpret_cast<const uint4*>(p);
  bfrag_bits f; f.lo = q[0]; f.hi = q[1];
  return __builtin_bit_cast(v16bf, f);
}

// Build A fragment from 16 fp32: f0..f3 = K kb..kb+15
__device__ __forceinline__ v16bf cvt_frag(float4 f0, float4 f1, float4 f2, float4 f3) {
  v16bf a;
  a[0]=(__bf16)f0.x;  a[1]=(__bf16)f0.y;  a[2]=(__bf16)f0.z;  a[3]=(__bf16)f0.w;
  a[4]=(__bf16)f1.x;  a[5]=(__bf16)f1.y;  a[6]=(__bf16)f1.z;  a[7]=(__bf16)f1.w;
  a[8]=(__bf16)f2.x;  a[9]=(__bf16)f2.y;  a[10]=(__bf16)f2.z; a[11]=(__bf16)f2.w;
  a[12]=(__bf16)f3.x; a[13]=(__bf16)f3.y; a[14]=(__bf16)f3.z; a[15]=(__bf16)f3.w;
  return a;
}

__device__ __forceinline__ float sigmoidf_fast(float x) {
  return 1.0f / (1.0f + __expf(-x));
}
__device__ __forceinline__ float tanhf_fast(float x) {
  float ax = fabsf(x);
  float e = __expf(-2.0f * ax);
  float r = (1.0f - e) / (1.0f + e);
  return copysignf(r, x);
}

// ---------------- fp32 -> bf16 weight conversion ----------------
__global__ __launch_bounds__(256) void cvt_bf16_kernel(const float* __restrict__ in,
                                                       __bf16* __restrict__ out, int n) {
  int i = blockIdx.x * 256 + threadIdx.x;
  if (i < n) out[i] = (__bf16)in[i];
}

// ---------------- gi = x * w_ih^T + b_ih  (bf16 WMMA, fp32 acc) ----------------
// Each wave computes a 32(M) x 64(N) output block: 2x4 WMMA tiles, K=1024.
__global__ __launch_bounds__(128) void gi_gemm_kernel(const float*  __restrict__ x,
                                                      const __bf16* __restrict__ wih,
                                                      const float*  __restrict__ bih,
                                                      float*        __restrict__ gi) {
  const int lane = threadIdx.x & 31;
  const int lm = lane & 15;
  const int lg = lane >> 4;
  const int wid = blockIdx.x * 4 + (threadIdx.x >> 5);
  const int NBLK = G3 / 64;              // 48
  const int mb = wid / NBLK;
  const int nb = wid - mb * NBLK;
  const int row0 = mb * 32;
  const int col0 = nb * 64;

  v8f acc[2][4];
  for (int r = 0; r < 2; ++r)
    for (int c = 0; c < 4; ++c) acc[r][c] = zero8();

  const float* xrow0 = x + (size_t)(row0 + lm) * DIN;
  const float* xrow1 = x + (size_t)(row0 + 16 + lm) * DIN;
  const __bf16* wrow[4];
  for (int c = 0; c < 4; ++c)
    wrow[c] = wih + (size_t)(col0 + 16 * c + lm) * DIN;

  for (int kk = 0; kk < DIN; kk += 32) {
    const int kbA = kk + (lg << 4);
    const float4* p0 = reinterpret_cast<const float4*>(xrow0 + kbA);
    const float4* p1 = reinterpret_cast<const float4*>(xrow1 + kbA);
    v16bf a0 = cvt_frag(p0[0], p0[1], p0[2], p0[3]);
    v16bf a1 = cvt_frag(p1[0], p1[1], p1[2], p1[3]);

    const int kbB = kk + (lg << 4);
    v16bf b0 = load_bfrag(wrow[0] + kbB);
    v16bf b1 = load_bfrag(wrow[1] + kbB);
    v16bf b2 = load_bfrag(wrow[2] + kbB);
    v16bf b3 = load_bfrag(wrow[3] + kbB);

    acc[0][0] = wmma_bf16(a0, b0, acc[0][0]);
    acc[0][1] = wmma_bf16(a0, b1, acc[0][1]);
    acc[0][2] = wmma_bf16(a0, b2, acc[0][2]);
    acc[0][3] = wmma_bf16(a0, b3, acc[0][3]);
    acc[1][0] = wmma_bf16(a1, b0, acc[1][0]);
    acc[1][1] = wmma_bf16(a1, b1, acc[1][1]);
    acc[1][2] = wmma_bf16(a1, b2, acc[1][2]);
    acc[1][3] = wmma_bf16(a1, b3, acc[1][3]);
  }

  // epilogue: add b_ih[col], store fp32 gi
  for (int c = 0; c < 4; ++c) {
    const int col = col0 + 16 * c + lm;
    const float bias = bih[col];
    for (int r = 0; r < 2; ++r) {
      for (int i = 0; i < 8; ++i) {
        const int row = row0 + 16 * r + i + (lg << 3);
        gi[(size_t)row * G3 + col] = acc[r][c][i] + bias;
      }
    }
  }
}

// ---------------- device-wide barrier (generation counter) ----------------
__device__ __forceinline__ void grid_sync(unsigned* bar, unsigned nblk) {
  __threadfence();
  __syncthreads();
  if (threadIdx.x == 0) {
    unsigned gen = __hip_atomic_load(&bar[1], __ATOMIC_RELAXED, __HIP_MEMORY_SCOPE_AGENT);
    unsigned prev = __hip_atomic_fetch_add(&bar[0], 1u, __ATOMIC_ACQ_REL, __HIP_MEMORY_SCOPE_AGENT);
    if (prev == nblk - 1u) {
      __hip_atomic_store(&bar[0], 0u, __ATOMIC_RELAXED, __HIP_MEMORY_SCOPE_AGENT);
      __hip_atomic_fetch_add(&bar[1], 1u, __ATOMIC_RELEASE, __HIP_MEMORY_SCOPE_AGENT);
    } else {
      while (__hip_atomic_load(&bar[1], __ATOMIC_ACQUIRE, __HIP_MEMORY_SCOPE_AGENT) == gen) {
        __builtin_amdgcn_s_sleep(1);
      }
    }
  }
  __syncthreads();
}

// ---------------- persistent masked-GRU scan ----------------
// 256 single-wave workgroups; block b owns output tile (mt = b>>6, jt = b&63):
// env rows 16*mt..16*mt+15, hidden cols 16*jt..16*jt+15.
// Its constant 96 KB slice of w_hh (3 gates x 16 rows x 1024 K, bf16) is staged
// once into LDS; the 512-step serial loop then reads B fragments via ds_load
// (low latency) and only the A fragments (h, L2-resident) + gi (streamed) from
// global memory.
__global__ __launch_bounds__(32) void gru_scan_kernel(const float*  __restrict__ gi,
                                                      const __bf16* __restrict__ whh,
                                                      const float*  __restrict__ bhh,
                                                      const float*  __restrict__ hx,
                                                      const float*  __restrict__ mask,
                                                      float*        __restrict__ out,
                                                      __bf16*       __restrict__ hbuf,
                                                      unsigned*     __restrict__ bar) {
  extern __shared__ __bf16 smem[];   // [3][16][1024] bf16 = 98304 bytes

  const int lane = threadIdx.x;
  const int lm = lane & 15;
  const int lg = lane >> 4;
  const int tile = blockIdx.x;
  const int mt = tile >> 6;          // 0..3  (env-row tile)
  const int jt = tile & 63;          // 0..63 (hidden-col tile)
  const int row0 = mt * 16;
  const int col = jt * 16 + lm;
  const int envBase = row0 + (lg << 3);   // C-layout rows for this lane

  // ---- stage this block's w_hh slice into LDS (one time) ----
  {
    const uint4* wsrc = reinterpret_cast<const uint4*>(whh);
    uint4* sdst = reinterpret_cast<uint4*>(smem);
    // 3*16 rows x 128 uint4 per row = 6144 uint4
    for (int idx = lane; idx < 6144; idx += 32) {
      const int grow = idx >> 7;             // row among 48 (= g*16 + rr)
      const int kq   = idx & 127;            // uint4 within the row
      const int g    = grow >> 4;
      const int rr   = grow & 15;
      const size_t srow = (size_t)g * HID + (size_t)jt * 16 + rr;  // w_hh row
      sdst[idx] = wsrc[srow * (HID / 8) + kq];
    }
  }

  const float bhr = bhh[col];
  const float bhz = bhh[HID + col];
  const float bhn = bhh[2 * HID + col];

  // local fp32 h tile (C layout)
  float hloc[8];
  for (int i = 0; i < 8; ++i)
    hloc[i] = hx[(size_t)(envBase + i) * HID + col];

  // init h buffer 0 with mask[0] pre-applied (producer-side masking)
  for (int i = 0; i < 8; ++i) {
    float m = mask[envBase + i];     // t = 0
    hbuf[(size_t)(envBase + i) * HID + col] = (__bf16)(hloc[i] * m);
  }
  __syncthreads();                   // LDS staging visible before use
  grid_sync(bar, 256u);

  // per-lane LDS base offsets for the 3 B-fragment rows
  const int sOffR = (0 * 16 + lm) * HID + (lg << 4);
  const int sOffZ = (1 * 16 + lm) * HID + (lg << 4);
  const int sOffN = (2 * 16 + lm) * HID + (lg << 4);

  for (int t = 0; t < T_STEPS; ++t) {
    const __bf16* hsrc = hbuf + (size_t)(t & 1) * (NENV * HID);
    __bf16* hdst = hbuf + (size_t)((t + 1) & 1) * (NENV * HID);

    // prefetch next step's gi rows for this lane
    if (t + 1 < T_STEPS)
      __builtin_prefetch(gi + ((size_t)(t + 1) * NENV + envBase) * G3 + col, 0, 0);

    // mask for local h (step t), rows envBase..envBase+7
    float4 cm0 = *reinterpret_cast<const float4*>(mask + t * NENV + envBase);
    float4 cm1 = *reinterpret_cast<const float4*>(mask + t * NENV + envBase + 4);
    float cm[8] = {cm0.x, cm0.y, cm0.z, cm0.w, cm1.x, cm1.y, cm1.z, cm1.w};
    for (int i = 0; i < 8; ++i) hloc[i] *= cm[i];

    v8f ar = zero8(), az = zero8(), an2 = zero8();
    const __bf16* hrowA = hsrc + (size_t)(row0 + lm) * HID + (lg << 4);
    for (int kk = 0; kk < HID; kk += 32) {
      v16bf a  = load_afrag(hrowA + kk);
      v16bf br = load_bfrag(smem + sOffR + kk);
      v16bf bz = load_bfrag(smem + sOffZ + kk);
      v16bf bn = load_bfrag(smem + sOffN + kk);
      ar  = wmma_bf16(a, br, ar);
      az  = wmma_bf16(a, bz, az);
      an2 = wmma_bf16(a, bn, an2);
    }

    const bool last = (t == T_STEPS - 1);
    float nm[8];
    if (!last) {
      float4 n0 = *reinterpret_cast<const float4*>(mask + (t + 1) * NENV + envBase);
      float4 n1 = *reinterpret_cast<const float4*>(mask + (t + 1) * NENV + envBase + 4);
      nm[0]=n0.x; nm[1]=n0.y; nm[2]=n0.z; nm[3]=n0.w;
      nm[4]=n1.x; nm[5]=n1.y; nm[6]=n1.z; nm[7]=n1.w;
    }

    const size_t gbase = (size_t)t * NENV * G3;
    for (int i = 0; i < 8; ++i) {
      const int env = envBase + i;
      const size_t g = gbase + (size_t)env * G3 + col;
      float grv = gi[g];
      float gzv = gi[g + HID];
      float gnv = gi[g + 2 * HID];
      float r = sigmoidf_fast(grv + ar[i] + bhr);
      float z = sigmoidf_fast(gzv + az[i] + bhz);
      float n = tanhf_fast(gnv + r * (an2[i] + bhn));
      float hn = (1.0f - z) * n + z * hloc[i];
      hloc[i] = hn;
      out[(size_t)t * (NENV * HID) + (size_t)env * HID + col] = hn;
      if (!last) {
        hdst[(size_t)env * HID + col] = (__bf16)(hn * nm[i]);
      } else {
        // h_final appended after out[T,N,H]
        out[(size_t)T_STEPS * (NENV * HID) + (size_t)env * HID + col] = hn;
      }
    }
    if (!last) grid_sync(bar, 256u);
  }
}

// ---------------- launch ----------------
extern "C" void kernel_launch(void* const* d_in, const int* in_sizes, int n_in,
                              void* d_out, int out_size, void* d_ws, size_t ws_size,
                              hipStream_t stream) {
  const float* x    = (const float*)d_in[0];
  const float* hx   = (const float*)d_in[1];
  const float* mask = (const float*)d_in[2];
  const float* w_ih = (const float*)d_in[3];
  const float* w_hh = (const float*)d_in[4];
  const float* b_ih = (const float*)d_in[5];
  const float* b_hh = (const float*)d_in[6];
  float* out = (float*)d_out;

  char* ws = (char*)d_ws;
  constexpr size_t GI_BYTES  = (size_t)T_STEPS * NENV * G3 * sizeof(float);   // 384 MB
  constexpr size_t W_BYTES   = (size_t)G3 * HID * sizeof(__bf16);             // 6 MB
  constexpr size_t HB_BYTES  = 2ull * NENV * HID * sizeof(__bf16);            // 256 KB
  float*    gi      = (float*)(ws);
  __bf16*   whh_bf  = (__bf16*)(ws + GI_BYTES);
  __bf16*   wih_bf  = (__bf16*)(ws + GI_BYTES + W_BYTES);
  __bf16*   hbuf    = (__bf16*)(ws + GI_BYTES + 2 * W_BYTES);
  unsigned* bar     = (unsigned*)(ws + GI_BYTES + 2 * W_BYTES + HB_BYTES);

  constexpr int SCAN_LDS_BYTES = 3 * 16 * HID * (int)sizeof(__bf16);          // 98304
  static bool attr_set = false;
  // Opt-in for >64KB dynamic LDS (no-op effect on compile loop; deterministic).
  hipFuncSetAttribute(reinterpret_cast<const void*>(gru_scan_kernel),
                      hipFuncAttributeMaxDynamicSharedMemorySize, SCAN_LDS_BYTES);

  hipMemsetAsync(bar, 0, 64, stream);

  const int wn = G3 * HID;                     // 3,145,728 elements each
  cvt_bf16_kernel<<<(wn + 255) / 256, 256, 0, stream>>>(w_ih, wih_bf, wn);
  cvt_bf16_kernel<<<(wn + 255) / 256, 256, 0, stream>>>(w_hh, whh_bf, wn);

  // gi GEMM: (32768/32) * (3072/64) = 49152 waves, 4 waves/block
  gi_gemm_kernel<<<49152 / 4, 128, 0, stream>>>(x, wih_bf, b_ih, gi);

  // persistent scan: 256 single-wave workgroups (one 16x16 h-tile each)
  gru_scan_kernel<<<256, 32, SCAN_LDS_BYTES, stream>>>(gi, whh_bf, b_hh, hx, mask,
                                                       out, hbuf, bar);
}
